// PCNN_88785563943309
// MI455X (gfx1250) — matrix operand order repaired and checked by reading, not compiled, not run on hardware
//
#include <hip/hip_runtime.h>

typedef _Float16 v16h __attribute__((ext_vector_type(16)));
typedef _Float16 v8h  __attribute__((ext_vector_type(8)));
typedef float    v8f  __attribute__((ext_vector_type(8)));
typedef int      v4i_ __attribute__((ext_vector_type(4)));

#define B_    2048
#define L_    120
#define WE_   50
#define PE_   5
#define D_    60
#define F_    230
#define K_    3
#define R_    53
#define FPP_  256   // F padded to 16*16 (even # of f-tiles -> 2-wide ILP loop)
#define CP_   192   // contraction: 3 taps * 64 (d padded 60->64) = 6 chunks of 32
#define ROWS_ 132   // 1 zero row + 120 emb rows + tail padding (need up to row 129)

// Hardware tanh on gfx1250 if the builtin exists, else libm.
#if __has_builtin(__builtin_amdgcn_tanhf)
#define TANH_(x) __builtin_amdgcn_tanhf(x)
#elif __has_builtin(__builtin_amdgcn_tanh_f32)
#define TANH_(x) __builtin_amdgcn_tanh_f32(x)
#else
#define TANH_(x) tanhf(x)
#endif

// CDNA5 async global->LDS copy path (ASYNCcnt), if this toolchain exposes it.
// Probe-revealed prototype: (v4i AS1* gsrc, v4i AS3* ldst, imm offset, imm cpol)
#if __has_builtin(__builtin_amdgcn_global_load_async_to_lds_b128) && \
    __has_builtin(__builtin_amdgcn_s_wait_asynccnt)
#define HAVE_ASYNC_LDS_ 1
typedef __attribute__((address_space(1))) v4i_* gas_ptr_;
typedef __attribute__((address_space(3))) v4i_* las_ptr_;
#endif

// order-preserving float<->int encoding so we can use integer LDS atomicMax
__device__ __forceinline__ int enc_f32(float v) {
    int i = __float_as_int(v);
    return (i >= 0) ? i : (i ^ 0x7FFFFFFF);
}
__device__ __forceinline__ float dec_f32(int i) {
    return __int_as_float((i >= 0) ? i : (i ^ 0x7FFFFFFF));
}

// Build WtT[FPP_][CP_] (f16): WtT[f][k*64+d] = conv_w[f,0,k,d], zero padded.
__global__ void pcnn_prep_w(const float* __restrict__ conv_w,
                            _Float16* __restrict__ WtT) {
    int idx = blockIdx.x * blockDim.x + threadIdx.x;
    if (idx >= FPP_ * CP_) return;
    int fp = idx / CP_;
    int c  = idx % CP_;
    int k = c >> 6, d = c & 63;
    float v = 0.f;
    if (fp < F_ && d < D_) v = conv_w[fp * (K_ * D_) + k * D_ + d];
    WtT[idx] = (_Float16)v;
}

__global__ __launch_bounds__(256) void pcnn_main(
    const int* __restrict__ sent, const int* __restrict__ pf,
    const int* __restrict__ epos,
    const float* __restrict__ W_word, const float* __restrict__ W_pos1,
    const float* __restrict__ W_pos2,
    const _Float16* __restrict__ WtT,
    const float* __restrict__ conv_b,
    const float* __restrict__ lin_w, const float* __restrict__ lin_b,
    float* __restrict__ out)
{
    __shared__ _Float16 embH[ROWS_][64];             // row 0 = zero pad (l = -1)
    __shared__ __align__(16) _Float16 wlds[32][CP_]; // B panel for 2 f-tiles (12KB)
    __shared__ int   pooled[3][FPP_];
    __shared__ float biasP[FPP_];

    const int b   = blockIdx.x;
    const int tid = threadIdx.x;

    for (int i = tid; i < 3 * FPP_; i += 256)
        ((int*)pooled)[i] = enc_f32(-2.0f);     // tanh in (-1,1), so -2 == -inf
    if (tid < FPP_) biasP[tid] = (tid < F_) ? conv_b[tid] : 0.f;

    // Gather embeddings into LDS as f16. Stored row r holds emb row r-1.
    for (int i = tid; i < ROWS_ * 64; i += 256) {
        int r = i >> 6, c = i & 63;
        float v = 0.f;
        if (r >= 1 && r <= L_) {
            int l = r - 1;
            if (c < WE_)              v = W_word[sent[b * L_ + l] * WE_ + c];
            else if (c < WE_ + PE_)   v = W_pos1[pf[b * 2 * L_ + l] * PE_ + (c - WE_)];
            else if (c < D_)          v = W_pos2[pf[b * 2 * L_ + L_ + l] * PE_ + (c - WE_ - PE_)];
        }
        embH[r][c] = (_Float16)v;
    }
    __syncthreads();

    const int p0 = epos[b * 2 + 0] + 1;   // segment boundaries
    const int p1 = epos[b * 2 + 1] + 1;

    const int wave = tid >> 5;            // l-tile 0..7 (rows wave*16 .. +15)
    const int ln   = tid & 31;
    const int mrow = ln & 15;             // A: M index / B,C: N index
    const int half = ln >> 4;

    // A fragments (16x32 f16) for all 6 K-chunks; chunk c: tap k=c/2, d-base=(c&1)*32.
    // 16-bit A 16x32 layout (contiguous k, matches B): lanes 0-15 K=base..+15; lanes 16-31 +16
    // -> two contiguous 16B LDS reads per chunk.
    v16h afr[6];
    #pragma unroll
    for (int c = 0; c < 6; ++c) {
        int k    = c >> 1;
        int base = (c & 1) * 32;
        int row  = wave * 16 + mrow + k;           // <= 129 < ROWS_
        v8h lo = *(const v8h*)&embH[row][base + half * 16];
        v8h hi = *(const v8h*)&embH[row][base + half * 16 + 8];
        v16h a;
        #pragma unroll
        for (int e = 0; e < 8; ++e) { a[e] = lo[e]; a[e + 8] = hi[e]; }
        afr[c] = a;
    }

    // tanh + piecewise max-pool for one 16x16 tile; branchless scalar maxes,
    // then 3 unconditional LDS atomic maxes (sentinel encodes below init).
    auto pool_tile = [&](const v8f& acc, int ft) {
        int   f    = ft * 16 + mrow;
        float bias = biasP[f];
        float m0 = -3.0f, m1 = -3.0f, m2 = -3.0f;
        #pragma unroll
        for (int r = 0; r < 8; ++r) {
            int  l     = wave * 16 + r + 8 * half;   // C/D: VGPR r -> M = r + 8*half
            bool valid = (l < L_);
            float v = TANH_(acc[r] + bias);
            v = valid ? v : -3.0f;
            bool s0 = l < p0;
            bool s2 = l >= p1;
            m0 = s0          ? fmaxf(m0, v) : m0;
            m1 = (!s0 && !s2)? fmaxf(m1, v) : m1;
            m2 = s2          ? fmaxf(m2, v) : m2;
        }
        atomicMax(&pooled[0][f], enc_f32(m0));
        atomicMax(&pooled[1][f], enc_f32(m1));
        atomicMax(&pooled[2][f], enc_f32(m2));
    };

    // Two f-tiles per iteration -> two independent WMMA accumulator chains.
    // The 12KB B panel is staged into LDS once per block-iteration so the
    // block reads WtT exactly once (instead of once per wave).
    for (int ft = 0; ft < 16; ft += 2) {
        __syncthreads();                    // wlds reuse: previous readers done
        const _Float16* src = WtT + ft * 16 * CP_;   // 32 rows x 192 halves = 12KB
#ifdef HAVE_ASYNC_LDS_
        #pragma unroll
        for (int j = 0; j < 3; ++j) {
            int idx = tid + j * 256;        // 768 chunks of 16B
            __builtin_amdgcn_global_load_async_to_lds_b128(
                (gas_ptr_)(src + idx * 8),
                (las_ptr_)(&wlds[0][0] + idx * 8), 0, 0);
        }
        __builtin_amdgcn_s_wait_asynccnt(0);
#else
        #pragma unroll
        for (int j = 0; j < 3; ++j) {
            int idx = tid + j * 256;
            *(float4*)(&wlds[0][0] + idx * 8) = *(const float4*)(src + idx * 8);
        }
#endif
        __syncthreads();

        v8f acc0 = {}, acc1 = {};
        const _Float16* wb0 = &wlds[mrow][half * 16];
        const _Float16* wb1 = &wlds[16 + mrow][half * 16];
        #pragma unroll
        for (int c = 0; c < 6; ++c) {
            v16h b0 = *(const v16h*)(wb0 + c * 32);
            v16h b1 = *(const v16h*)(wb1 + c * 32);
            acc0 = __builtin_amdgcn_wmma_f32_16x16x32_f16(
                false, afr[c], false, b0, (short)0, acc0, false, false);
            acc1 = __builtin_amdgcn_wmma_f32_16x16x32_f16(
                false, afr[c], false, b1, (short)0, acc1, false, false);
        }
        pool_tile(acc0, ft);
        pool_tile(acc1, ft + 1);
    }
    __syncthreads();

    // feats[f*3+s] @ lin_w[R, 3F]^T + lin_b ; tiny, 53 threads do full dots.
    if (tid < R_) {
        float sum = lin_b[tid];
        for (int f = 0; f < F_; ++f) {
            #pragma unroll
            for (int s = 0; s < 3; ++s)
                sum += dec_f32(pooled[s][f]) * lin_w[tid * (3 * F_) + f * 3 + s];
        }
        out[b * R_ + tid] = sum;
    }
}

extern "C" void kernel_launch(void* const* d_in, const int* in_sizes, int n_in,
                              void* d_out, int out_size, void* d_ws, size_t ws_size,
                              hipStream_t stream) {
    const int*   sent   = (const int*)  d_in[0];
    const int*   pf     = (const int*)  d_in[1];
    const int*   epos   = (const int*)  d_in[2];
    const float* W_word = (const float*)d_in[3];
    const float* W_pos1 = (const float*)d_in[4];
    const float* W_pos2 = (const float*)d_in[5];
    const float* conv_w = (const float*)d_in[6];
    const float* conv_b = (const float*)d_in[7];
    const float* lin_w  = (const float*)d_in[8];
    const float* lin_b  = (const float*)d_in[9];
    float* out = (float*)d_out;

    _Float16* WtT = (_Float16*)d_ws;   // FPP_*CP_*2 = 98304 bytes

    int prep_elems = FPP_ * CP_;
    pcnn_prep_w<<<(prep_elems + 255) / 256, 256, 0, stream>>>(conv_w, WtT);

    pcnn_main<<<B_, 256, 0, stream>>>(sent, pf, epos, W_word, W_pos1, W_pos2,
                                      WtT, conv_b, lin_w, lin_b, out);
}
